// MultiheadAtt_962072674420
// MI455X (gfx1250) — compile-verified, matrix operand order checked by reading, not run on hardware
//
#include <hip/hip_runtime.h>
#include <stdint.h>

// ---------------------------------------------------------------------------
// MultiheadAtt for MI455X (gfx1250): bf16 WMMA, double-buffered LDS staging,
// async global->LDS (ASYNCcnt) for bf16 operands, branch-free epilogues.
// B=2, S=2048, D=1024, H=16, HD=64
// ---------------------------------------------------------------------------

#define BB 2
#define SS 2048
#define DD 1024
#define HH 16
#define HDD 64

typedef __attribute__((ext_vector_type(16))) __bf16 v16bf;
typedef __attribute__((ext_vector_type(8)))  __bf16 v8bf;
typedef __attribute__((ext_vector_type(8)))  float  v8f;

#define WMMA_BF16(a, b, c) \
  __builtin_amdgcn_wmma_f32_16x16x32_bf16(false, (a), false, (b), (short)0, (c), false, false)

// --- CDNA5 async global->LDS (tracked by ASYNCcnt) --------------------------
__device__ __forceinline__ uint32_t lds_off32(const void* p) {
  // generic pointers into the LDS aperture carry the LDS byte offset in [31:0]
  return (uint32_t)(uintptr_t)p;
}
__device__ __forceinline__ void async_b128(uint32_t lds_off, const void* gaddr) {
  asm volatile("global_load_async_to_lds_b128 %0, %1, off"
               :: "v"(lds_off), "v"(gaddr) : "memory");
}
__device__ __forceinline__ void wait_async0() {
  asm volatile("s_wait_asynccnt 0" ::: "memory");
}

// --- fragment load from an LDS tile row -------------------------------------
// p points at rowbase + half*8 (bf16 elems); lane's 16 values: p[0..7], p[16..23]
__device__ __forceinline__ v16bf frag_ld(const __bf16* p) {
  v16bf r;
#pragma unroll
  for (int i = 0; i < 8; ++i) { r[i] = p[i]; r[i + 8] = p[i + 16]; }
  return r;
}

// ---------------------------------------------------------------------------
// Kernel 1: QKV projections, x @ W^T + b.  Block tile 128x128, K step 32,
// double-buffered LDS.  Each wave: 2x4 grid of 16x16 WMMA tiles.
// grid = (256, 1, 3)
// ---------------------------------------------------------------------------
__global__ void __launch_bounds__(256) k_proj_qkv(
    const float* __restrict__ q_in, const float* __restrict__ k_in,
    const float* __restrict__ v_in,
    const float* __restrict__ Wq, const float* __restrict__ bq,
    const float* __restrict__ Wk, const float* __restrict__ bk,
    const float* __restrict__ Wv, const float* __restrict__ bv,
    __bf16* __restrict__ qbf,   // [B,H,S,64] pre-scaled 1/8
    __bf16* __restrict__ kbf,   // [B,H,S,64]
    __bf16* __restrict__ vt)    // [B,H,64,S]
{
  __shared__ __align__(32) __bf16 sA[2][128 * 32];
  __shared__ __align__(32) __bf16 sB[2][128 * 32];

  const int tid = threadIdx.x;
  const int wid = tid >> 5, lane = tid & 31;
  const int n16 = lane & 15, half = lane >> 4;

  const int p = blockIdx.z;
  const float* X; const float* W; const float* bias;
  if (p == 0)      { X = q_in; W = Wq; bias = bq; }
  else if (p == 1) { X = k_in; W = Wk; bias = bk; }
  else             { X = v_in; W = Wv; bias = bv; }

  const int mBase = (blockIdx.x >> 3) * 128;
  const int nBase = (blockIdx.x & 7) * 128;
  const int wm = wid & 3;                      // 32-row group
  const int wn = wid >> 2;                     // 64-col group

  const int srow = tid >> 1;                   // staging row 0..127
  const int skh  = (tid & 1) * 16;             // staging k-half (16 elems)
  const float* gArow = X + (size_t)(mBase + srow) * DD + skh;
  const float* gBrow = W + (size_t)(nBase + srow) * DD + skh;

  float ra[16], rb[16];
  auto gload = [&](int k0) {
#pragma unroll
    for (int i = 0; i < 16; ++i) { ra[i] = gArow[k0 + i]; rb[i] = gBrow[k0 + i]; }
  };
  auto sstore = [&](int buf) {
    v16bf ta, tb;
#pragma unroll
    for (int i = 0; i < 16; ++i) { ta[i] = (__bf16)ra[i]; tb[i] = (__bf16)rb[i]; }
    *(v16bf*)&sA[buf][srow * 32 + skh] = ta;
    *(v16bf*)&sB[buf][srow * 32 + skh] = tb;
  };

  v8f zero = {};
  v8f c[2][4];
#pragma unroll
  for (int it = 0; it < 2; ++it)
#pragma unroll
    for (int jt = 0; jt < 4; ++jt) c[it][jt] = zero;

  gload(0); sstore(0);
  __syncthreads();

  int buf = 0;
  for (int k0 = 0; k0 < DD; k0 += 32) {
    const bool more = (k0 + 32) < DD;
    if (more) gload(k0 + 32);                  // overlap with WMMAs below

    v16bf af[2], bfr[4];
#pragma unroll
    for (int it = 0; it < 2; ++it)
      af[it] = frag_ld(&sA[buf][(wm * 32 + it * 16 + n16) * 32 + half * 8]);
#pragma unroll
    for (int jt = 0; jt < 4; ++jt)
      bfr[jt] = frag_ld(&sB[buf][(wn * 64 + jt * 16 + n16) * 32 + half * 8]);
#pragma unroll
    for (int it = 0; it < 2; ++it)
#pragma unroll
      for (int jt = 0; jt < 4; ++jt)
        c[it][jt] = WMMA_BF16(af[it], bfr[jt], c[it][jt]);

    if (more) sstore(buf ^ 1);                 // s_wait_loadcnt lands here
    __syncthreads();
    buf ^= 1;
  }

  // ---- epilogue: single projection branch, vectorized where contiguous ----
  if (p == 2) {
    // V transposed [B,H,64,S]: a lane's 8 acc elems are 8 consecutive s values
#pragma unroll
    for (int jt = 0; jt < 4; ++jt) {
      const int j = nBase + wn * 64 + jt * 16 + n16;
      const float bn = bias[j];
      const int h = j >> 6, hd = j & 63;
#pragma unroll
      for (int it = 0; it < 2; ++it) {
        const int i0 = mBase + wm * 32 + it * 16 + 8 * half;  // multiple of 8
        const int bi = i0 >> 11, s0 = i0 & 2047;
        v8bf vv;
#pragma unroll
        for (int r = 0; r < 8; ++r) vv[r] = (__bf16)(c[it][jt][r] + bn);
        *(v8bf*)&vt[(((size_t)bi * HH + h) * HDD + hd) * SS + s0] = vv;
      }
    }
  } else {
    __bf16* dst = (p == 0) ? qbf : kbf;
    const float sc = (p == 0) ? 0.125f : 1.0f;
#pragma unroll
    for (int jt = 0; jt < 4; ++jt) {
      const int j = nBase + wn * 64 + jt * 16 + n16;
      const float bn = bias[j];
      const int h = j >> 6, hd = j & 63;
#pragma unroll
      for (int it = 0; it < 2; ++it) {
        const int i0 = mBase + wm * 32 + it * 16 + 8 * half;
        const int bi = i0 >> 11, s0 = i0 & 2047;
        __bf16* base = dst + (((size_t)bi * HH + h) * SS + s0) * HDD + hd;
#pragma unroll
        for (int r = 0; r < 8; ++r)
          base[(size_t)r * HDD] = (__bf16)((c[it][jt][r] + bn) * sc);
      }
    }
  }
}

// ---------------------------------------------------------------------------
// Kernel 2: scores = (Q/8) @ K^T per (b,h).  128x128 tile; Q/K fully
// async-loaded to LDS once (K-dim = 64).  grid = 8192 blocks.
// ---------------------------------------------------------------------------
__global__ void __launch_bounds__(256) k_scores(
    const __bf16* __restrict__ qbf, const __bf16* __restrict__ kbf,
    float* __restrict__ probs)
{
  __shared__ __align__(32) __bf16 sQ[128 * 64];
  __shared__ __align__(32) __bf16 sK[128 * 64];

  const int tid = threadIdx.x;
  const int wid = tid >> 5, lane = tid & 31;
  const int n16 = lane & 15, half = lane >> 4;

  const int bh  = blockIdx.x >> 8;
  const int t   = blockIdx.x & 255;
  const int qb0 = (t >> 4) * 128;
  const int kb0 = (t & 15) * 128;

  const __bf16* qb = qbf + (size_t)bh * SS * HDD;
  const __bf16* kb = kbf + (size_t)bh * SS * HDD;

#pragma unroll
  for (int i = 0; i < 4; ++i) {
    const int lin = i * 256 + tid;
    const int row = lin >> 3, seg = (lin & 7) * 8;
    async_b128(lds_off32(&sQ[row * 64 + seg]), qb + (size_t)(qb0 + row) * HDD + seg);
    async_b128(lds_off32(&sK[row * 64 + seg]), kb + (size_t)(kb0 + row) * HDD + seg);
  }
  wait_async0();
  __syncthreads();

  const int wm = wid & 3;
  const int wn = wid >> 2;

  v8f zero = {};
  v8f c[2][4];
#pragma unroll
  for (int it = 0; it < 2; ++it)
#pragma unroll
    for (int jt = 0; jt < 4; ++jt) c[it][jt] = zero;

#pragma unroll
  for (int k0 = 0; k0 < HDD; k0 += 32) {
    v16bf af[2], bfr[4];
#pragma unroll
    for (int it = 0; it < 2; ++it)
      af[it] = frag_ld(&sQ[(wm * 32 + it * 16 + n16) * 64 + k0 + half * 8]);
#pragma unroll
    for (int jt = 0; jt < 4; ++jt)
      bfr[jt] = frag_ld(&sK[(wn * 64 + jt * 16 + n16) * 64 + k0 + half * 8]);
#pragma unroll
    for (int it = 0; it < 2; ++it)
#pragma unroll
      for (int jt = 0; jt < 4; ++jt)
        c[it][jt] = WMMA_BF16(af[it], bfr[jt], c[it][jt]);
  }

  float* prow = probs + (size_t)bh * SS * SS;
#pragma unroll
  for (int it = 0; it < 2; ++it) {
#pragma unroll
    for (int jt = 0; jt < 4; ++jt) {
#pragma unroll
      for (int r = 0; r < 8; ++r) {
        const int row = qb0 + wm * 32 + it * 16 + r + 8 * half;
        const int col = kb0 + wn * 64 + jt * 16 + n16;
        prow[(size_t)row * SS + col] = c[it][jt][r];
      }
    }
  }
}

// ---------------------------------------------------------------------------
// Kernel 3: row softmax over probs in place. One block per row.
// ---------------------------------------------------------------------------
__global__ void __launch_bounds__(256) k_softmax(float* __restrict__ probs)
{
  __shared__ float smax[8];
  __shared__ float ssum[8];

  const int tid = threadIdx.x;
  const int wid = tid >> 5, lane = tid & 31;
  float* row = probs + (size_t)blockIdx.x * SS;

  float x[8];
#pragma unroll
  for (int i = 0; i < 8; ++i) x[i] = row[tid + i * 256];

  float m = x[0];
#pragma unroll
  for (int i = 1; i < 8; ++i) m = fmaxf(m, x[i]);
#pragma unroll
  for (int o = 16; o > 0; o >>= 1) m = fmaxf(m, __shfl_xor(m, o, 32));
  if (lane == 0) smax[wid] = m;
  __syncthreads();
  m = smax[0];
#pragma unroll
  for (int i = 1; i < 8; ++i) m = fmaxf(m, smax[i]);

  float s = 0.f;
#pragma unroll
  for (int i = 0; i < 8; ++i) { x[i] = __expf(x[i] - m); s += x[i]; }
#pragma unroll
  for (int o = 16; o > 0; o >>= 1) s += __shfl_xor(s, o, 32);
  if (lane == 0) ssum[wid] = s;
  __syncthreads();
  s = 0.f;
#pragma unroll
  for (int i = 0; i < 8; ++i) s += ssum[i];

  const float inv = 1.0f / s;
#pragma unroll
  for (int i = 0; i < 8; ++i) row[tid + i * 256] = x[i] * inv;
}

// ---------------------------------------------------------------------------
// Kernel 4: out_heads = P @ V per (b,h).  128 rows x 64 cols per block.
// Double-buffered: P (f32) cvt-staged, V^T (bf16) async-staged.
// grid = 512 blocks; wave w owns rows 16w, all 4 N tiles.
// ---------------------------------------------------------------------------
__global__ void __launch_bounds__(256) k_pv(
    const float* __restrict__ probs, const __bf16* __restrict__ vt,
    __bf16* __restrict__ abf)
{
  __shared__ __align__(32) __bf16 sP[2][128 * 32];
  __shared__ __align__(32) __bf16 sV[2][64 * 32];

  const int tid = threadIdx.x;
  const int wid = tid >> 5, lane = tid & 31;
  const int n16 = lane & 15, half = lane >> 4;

  const int bh   = blockIdx.x >> 4;
  const int q0   = (blockIdx.x & 15) * 128;

  const float*  pr = probs + (size_t)bh * SS * SS;
  const __bf16* vb = vt    + (size_t)bh * HDD * SS;

  const int srow = tid >> 1;                   // P staging row 0..127
  const int skh  = (tid & 1) * 16;
  const int vrow = tid >> 2;                   // V staging row 0..63
  const int vseg = (tid & 3) * 8;
  const float* gProw = pr + (size_t)(q0 + srow) * SS + skh;

  float rp[16];
  auto gloadP = [&](int k0) {
#pragma unroll
    for (int i = 0; i < 16; ++i) rp[i] = gProw[k0 + i];
  };
  auto sstoreP = [&](int b) {
    v16bf t;
#pragma unroll
    for (int i = 0; i < 16; ++i) t[i] = (__bf16)rp[i];
    *(v16bf*)&sP[b][srow * 32 + skh] = t;
  };
  auto asyncV = [&](int b, int k0) {
    async_b128(lds_off32(&sV[b][vrow * 32 + vseg]),
               vb + (size_t)vrow * SS + k0 + vseg);
  };

  v8f zero = {};
  v8f c[4];
#pragma unroll
  for (int jt = 0; jt < 4; ++jt) c[jt] = zero;

  asyncV(0, 0); gloadP(0); sstoreP(0);
  wait_async0();
  __syncthreads();

  int buf = 0;
  for (int k0 = 0; k0 < SS; k0 += 32) {
    const bool more = (k0 + 32) < SS;
    if (more) { asyncV(buf ^ 1, k0 + 32); gloadP(k0 + 32); }

    v16bf a = frag_ld(&sP[buf][(wid * 16 + n16) * 32 + half * 8]);
    v16bf bfr[4];
#pragma unroll
    for (int jt = 0; jt < 4; ++jt)
      bfr[jt] = frag_ld(&sV[buf][(jt * 16 + n16) * 32 + half * 8]);
#pragma unroll
    for (int jt = 0; jt < 4; ++jt) c[jt] = WMMA_BF16(a, bfr[jt], c[jt]);

    if (more) sstoreP(buf ^ 1);
    wait_async0();
    __syncthreads();
    buf ^= 1;
  }

  const int bi = bh >> 4, h = bh & 15;
#pragma unroll
  for (int jt = 0; jt < 4; ++jt) {
    const int d = jt * 16 + n16;
#pragma unroll
    for (int r = 0; r < 8; ++r) {
      const int s = q0 + wid * 16 + r + 8 * half;
      abf[((size_t)bi * SS + s) * DD + h * HDD + d] = (__bf16)c[jt][r];
    }
  }
}

// ---------------------------------------------------------------------------
// Kernel 5: out = attn @ Wo^T + bo -> f32.  128x128 block tile, double
// buffered: A (bf16) async-staged, B (Wo, f32) cvt-staged.  grid = 256.
// ---------------------------------------------------------------------------
__global__ void __launch_bounds__(256) k_oproj(
    const __bf16* __restrict__ abf, const float* __restrict__ Wo,
    const float* __restrict__ bo, float* __restrict__ out)
{
  __shared__ __align__(32) __bf16 sA[2][128 * 32];
  __shared__ __align__(32) __bf16 sB[2][128 * 32];

  const int tid = threadIdx.x;
  const int wid = tid >> 5, lane = tid & 31;
  const int n16 = lane & 15, half = lane >> 4;

  const int mBase = (blockIdx.x >> 3) * 128;
  const int nBase = (blockIdx.x & 7) * 128;
  const int wm = wid & 3, wn = wid >> 2;

  const int srow = tid >> 1;
  const int skh  = (tid & 1) * 16;
  const float* gBrow = Wo + (size_t)(nBase + srow) * DD + skh;

  float rb[16];
  auto gloadB = [&](int k0) {
#pragma unroll
    for (int i = 0; i < 16; ++i) rb[i] = gBrow[k0 + i];
  };
  auto sstoreB = [&](int b) {
    v16bf t;
#pragma unroll
    for (int i = 0; i < 16; ++i) t[i] = (__bf16)rb[i];
    *(v16bf*)&sB[b][srow * 32 + skh] = t;
  };
  auto asyncA = [&](int b, int k0) {
#pragma unroll
    for (int j = 0; j < 2; ++j) {
      const int lin = j * 256 + tid;
      const int row = lin >> 2, seg = (lin & 3) * 8;
      async_b128(lds_off32(&sA[b][row * 32 + seg]),
                 abf + (size_t)(mBase + row) * DD + k0 + seg);
    }
  };

  v8f zero = {};
  v8f c[2][4];
#pragma unroll
  for (int it = 0; it < 2; ++it)
#pragma unroll
    for (int jt = 0; jt < 4; ++jt) c[it][jt] = zero;

  asyncA(0, 0); gloadB(0); sstoreB(0);
  wait_async0();
  __syncthreads();

  int buf = 0;
  for (int k0 = 0; k0 < DD; k0 += 32) {
    const bool more = (k0 + 32) < DD;
    if (more) { asyncA(buf ^ 1, k0 + 32); gloadB(k0 + 32); }

    v16bf af[2], bfr[4];
#pragma unroll
    for (int it = 0; it < 2; ++it)
      af[it] = frag_ld(&sA[buf][(wm * 32 + it * 16 + n16) * 32 + half * 8]);
#pragma unroll
    for (int jt = 0; jt < 4; ++jt)
      bfr[jt] = frag_ld(&sB[buf][(wn * 64 + jt * 16 + n16) * 32 + half * 8]);
#pragma unroll
    for (int it = 0; it < 2; ++it)
#pragma unroll
      for (int jt = 0; jt < 4; ++jt)
        c[it][jt] = WMMA_BF16(af[it], bfr[jt], c[it][jt]);

    if (more) sstoreB(buf ^ 1);
    wait_async0();
    __syncthreads();
    buf ^= 1;
  }

#pragma unroll
  for (int jt = 0; jt < 4; ++jt) {
    const int col = nBase + wn * 64 + jt * 16 + n16;
    const float bn = bo[col];
#pragma unroll
    for (int it = 0; it < 2; ++it) {
#pragma unroll
      for (int r = 0; r < 8; ++r) {
        const int i = mBase + wm * 32 + it * 16 + r + 8 * half;
        out[(size_t)i * DD + col] = c[it][jt][r] + bn;
      }
    }
  }
}

// ---------------------------------------------------------------------------
extern "C" void kernel_launch(void* const* d_in, const int* in_sizes, int n_in,
                              void* d_out, int out_size, void* d_ws, size_t ws_size,
                              hipStream_t stream) {
  const float* query  = (const float*)d_in[0];
  const float* key_in = (const float*)d_in[1];
  const float* value  = (const float*)d_in[2];
  const float* Wq = (const float*)d_in[3];  const float* bq = (const float*)d_in[4];
  const float* Wk = (const float*)d_in[5];  const float* bk = (const float*)d_in[6];
  const float* Wv = (const float*)d_in[7];  const float* bv = (const float*)d_in[8];
  const float* Wo = (const float*)d_in[9];  const float* bo = (const float*)d_in[10];

  float* out   = (float*)d_out;
  float* probs = out + (size_t)BB * SS * DD;          // [B*H, S, S]

  const size_t NQKV = (size_t)BB * SS * DD;
  __bf16* qbf = (__bf16*)d_ws;
  __bf16* kbf = qbf + NQKV;
  __bf16* vt  = kbf + NQKV;
  __bf16* abf = vt  + NQKV;                           // 32 MB bf16 scratch

  k_proj_qkv<<<dim3(256, 1, 3), 256, 0, stream>>>(
      query, key_in, value, Wq, bq, Wk, bk, Wv, bv, qbf, kbf, vt);
  k_scores<<<8192, 256, 0, stream>>>(qbf, kbf, probs);
  k_softmax<<<BB * HH * SS, 256, 0, stream>>>(probs);
  k_pv<<<512, 256, 0, stream>>>(probs, vt, abf);
  k_oproj<<<256, 256, 0, stream>>>(abf, Wo, bo, out);
}
